// BCL_12352325943483
// MI455X (gfx1250) — matrix operand order repaired and checked by reading, not compiled, not run on hardware
//
#include <hip/hip_runtime.h>

typedef __attribute__((ext_vector_type(16))) _Float16 v16h;
typedef __attribute__((ext_vector_type(8)))  float    v8f;
typedef __attribute__((ext_vector_type(4)))  float    v4f;

constexpr int NGRID = 128;                 // reference N=128
constexpr int NG2   = NGRID * NGRID;       // 16384
constexpr int NG3   = NGRID * NG2;         // 2097152 cells
constexpr int CH    = 32;                  // channels in/out
constexpr int NTAPS = 27;                  // 3x3x3

// Workspace layout (bytes). Grid is channel-last f32: 128^3 cells * 32 ch = 256 MiB.
constexpr size_t OFF_PARAMS  = 0;          // float[16]: spacing[3], lo_edge[3], hi_edge[3]
constexpr size_t OFF_CELLS   = 256;        // int[3*P]
constexpr size_t OFF_SPLATW  = 0x8000;     // float[8*P]
constexpr size_t OFF_ROWCELL = 0x18000;    // int[8*P]   packed gather cell (x<<14|y<<7|z)
constexpr size_t OFF_ROWW    = 0x28000;    // float[8*P] gather weight (dist[4i+2jk]/8)
constexpr size_t OFF_BPACK   = 0x38000;    // _Float16[27*2*32*16] B in WMMA lane layout
constexpr size_t OFF_GRID    = 0x50000;    // float[128^3 * 32], channel-last

// ---------------- Stage 1: min/max -> spacing, lo_edge, hi_edge ----------------
__global__ void k_params(const float* __restrict__ data, int P,
                         float* __restrict__ params) {
  __shared__ float smn[3][256], smx[3][256];
  int t = threadIdx.x;
  float mn[3] = {1e30f, 1e30f, 1e30f}, mx[3] = {-1e30f, -1e30f, -1e30f};
  for (int p = t; p < P; p += 256)
    for (int d = 0; d < 3; ++d) {
      float v = data[p * 3 + d];
      mn[d] = fminf(mn[d], v);
      mx[d] = fmaxf(mx[d], v);
    }
  for (int d = 0; d < 3; ++d) { smn[d][t] = mn[d]; smx[d][t] = mx[d]; }
  __syncthreads();
  for (int s = 128; s > 0; s >>= 1) {
    if (t < s)
      for (int d = 0; d < 3; ++d) {
        smn[d][t] = fminf(smn[d][t], smn[d][t + s]);
        smx[d][t] = fmaxf(smx[d][t], smx[d][t + s]);
      }
    __syncthreads();
  }
  if (t == 0)
    for (int d = 0; d < 3; ++d) {
      float mind = floorf(smn[d][0]);
      float maxd = ceilf(smx[d][0]);
      float sp   = (maxd - mind) / (float)(NGRID - 9);
      params[d]     = sp;
      params[3 + d] = mind - 4.0f * sp;   // lo_edge
      params[6 + d] = maxd + 4.0f * sp;   // hi_edge
    }
}

// ---------------- Stage 2: per-point cells, dists, weights ----------------
__global__ void k_point_setup(const float* __restrict__ data, int P,
                              const float* __restrict__ params,
                              int* __restrict__ cells, float* __restrict__ splatw,
                              int* __restrict__ rowcell, float* __restrict__ roww) {
  int p = blockIdx.x * blockDim.x + threadIdx.x;
  if (p >= P) return;
  float sp[3], lo[3], hi[3];
  for (int d = 0; d < 3; ++d) { sp[d] = params[d]; lo[d] = params[3 + d]; hi[d] = params[6 + d]; }
  int c[3]; float glo[3], ghi[3], x[3];
  const float invN1 = 1.0f / (float)(NGRID - 1);
  for (int d = 0; d < 3; ++d) {
    x[d] = data[p * 3 + d];
    c[d] = (int)floorf((x[d] - lo[d]) / sp[d]);
    float alo = lo[d] + (hi[d] - lo[d]) * ((float)c[d] * invN1);
    float ahi = lo[d] + (hi[d] - lo[d]) * ((float)(c[d] + 1) * invN1);
    glo[d] = (float)(int)alo;   // reference: axes.astype(int32).astype(f32)
    ghi[d] = (float)(int)ahi;
    cells[p * 3 + d] = c[d];
  }
  float dist[8];
  for (int cn = 0; cn < 8; ++cn) {
    int i = (cn >> 2) & 1, j = (cn >> 1) & 1, k = cn & 1;
    float cx = i ? ghi[0] : glo[0];
    float cy = j ? ghi[1] : glo[1];
    float cz = k ? ghi[2] : glo[2];
    float dx = x[0] - cx, dy = x[1] - cy, dz = x[2] - cz;
    dist[cn] = sqrtf(dx * dx + dy * dy + dz * dz);
  }
  for (int cn = 0; cn < 8; ++cn) {
    int i = (cn >> 2) & 1, j = (cn >> 1) & 1, k = cn & 1;
    splatw[p * 8 + cn] = 1.0f / dist[cn];                 // splat: feat / dist[4i+2j+k]
    int wcol = 4 * i + 2 * j * k;                         // reference quirk: 2*j*k
    roww[p * 8 + cn] = dist[wcol] * 0.125f;               // gather weight, /8 folded in
    int gx = c[0] + i, gy = c[1] + j, gz = c[2] + k;
    rowcell[p * 8 + cn] = (gx << 14) | (gy << 7) | gz;
  }
}

// ---------------- Stage 3: splat into channel-last grid (coalesced) ----------------
__global__ void k_splat(const float* __restrict__ feat, int P,
                        const int* __restrict__ cells, const float* __restrict__ splatw,
                        float* __restrict__ grid) {
  int p = blockIdx.x;
  int ch = threadIdx.x;                       // 0..31
  float f = feat[ch * P + p];
  int cx = cells[p * 3], cy = cells[p * 3 + 1], cz = cells[p * 3 + 2];
  unsigned base = (unsigned)(cx * NG2 + cy * NGRID + cz) * CH;
  #pragma unroll
  for (int cn = 0; cn < 8; ++cn) {
    int i = (cn >> 2) & 1, j = (cn >> 1) & 1, k = cn & 1;
    unsigned off = base + (unsigned)(i * NG2 + j * NGRID + k) * CH + ch;
    atomicAdd(grid + off, f * splatw[p * 8 + cn]);        // 128B coalesced per corner
  }
}

// ------- Stage 4: pack conv weights to f16 directly in WMMA B lane layout -------
// bp[((t*2 + ntile)*32 + lane)*16 + h] = w[n][ci][t],
//   n = ntile*16 + (lane&15), ci = (h<8 ? 0 : 8) + (lane>>4)*8 + h   (B 32x16 f16 layout)
__global__ void k_bpack(const float* __restrict__ w, _Float16* __restrict__ bp) {
  int idx = blockIdx.x * blockDim.x + threadIdx.x;
  if (idx >= NTAPS * 2 * 32 * 16) return;
  int h    = idx & 15;
  int lane = (idx >> 4) & 31;
  int nt   = (idx >> 9) & 1;
  int t    = idx >> 10;
  int n  = nt * 16 + (lane & 15);
  int ci = (h < 8 ? 0 : 8) + ((lane >> 4) & 1) * 8 + h;
  bp[idx] = (_Float16)w[(n * CH + ci) * NTAPS + t];
}

// ---------------- Stage 5: sparse conv + gather via WMMA ----------------
// M = 8*P rows (point x corner), N = 32 out-channels (two 16-tiles), K = 27*32.
// A tile 16x32 f16 per lane: a[0..7] = grid[ci=laneHi*8+0..7],
//                            a[8..15] = grid[ci=16+laneHi*8+0..7]  -> 4x b128 loads.
__global__ void k_conv_gather(const float* __restrict__ grid,
                              const _Float16* __restrict__ bp,
                              const float* __restrict__ bias,
                              const int* __restrict__ rowcell,
                              const float* __restrict__ roww,
                              int rows, int P, float* __restrict__ out) {
  int wave = (int)(threadIdx.x >> 5);
  int lane = (int)(threadIdx.x & 31);
  int tile = blockIdx.x * (int)(blockDim.x >> 5) + wave;
  int ntiles = (rows + 15) >> 4;
  if (tile >= ntiles) return;               // uniform per wave: EXEC stays all-ones

  int row0 = tile * 16;
  int mrow = row0 + (lane & 15);            // A row this lane feeds
  int mrowC = mrow < rows ? mrow : rows - 1;
  int cellp = rowcell[mrowC];
  int gx = (cellp >> 14) & 127, gy = (cellp >> 7) & 127, gz = cellp & 127;
  int laneHi = (lane >> 4) & 1;
  unsigned cellbase = (unsigned)(gx * NG2 + gy * NGRID + gz) * CH;

  v8f acc0 = {}; v8f acc1 = {};
  for (int t = 0; t < NTAPS; ++t) {
    int dx = t / 9, dy = (t / 3) % 3, dz = t % 3;
    const float* gptr = grid + cellbase + (unsigned)(dx * NG2 + dy * NGRID + dz) * CH;
    if (t + 1 < NTAPS) {                    // gfx1250 global_prefetch_b8 for next tap
      int t1 = t + 1;
      __builtin_prefetch(grid + cellbase +
                         (unsigned)((t1 / 9) * NG2 + ((t1 / 3) % 3) * NGRID + (t1 % 3)) * CH, 0, 3);
    }
    // A fragment: 4x contiguous float4 loads, cvt to f16
    v4f f0 = *(const v4f*)(gptr + laneHi * 8);
    v4f f1 = *(const v4f*)(gptr + laneHi * 8 + 4);
    v4f f2 = *(const v4f*)(gptr + 16 + laneHi * 8);
    v4f f3 = *(const v4f*)(gptr + 16 + laneHi * 8 + 4);
    v16h a;
    #pragma unroll
    for (int q = 0; q < 4; ++q) {
      a[q]      = (_Float16)f0[q];
      a[4 + q]  = (_Float16)f1[q];
      a[8 + q]  = (_Float16)f2[q];
      a[12 + q] = (_Float16)f3[q];
    }
    // B fragments: whole v16h per lane, contiguous (packed in register layout)
    v16h b0 = *(const v16h*)(bp + ((size_t)(t * 2 + 0) * 32 + lane) * 16);
    v16h b1 = *(const v16h*)(bp + ((size_t)(t * 2 + 1) * 32 + lane) * 16);
    acc0 = __builtin_amdgcn_wmma_f32_16x16x32_f16(false, a, false, b0, (short)0, acc0, false, false);
    acc1 = __builtin_amdgcn_wmma_f32_16x16x32_f16(false, a, false, b1, (short)0, acc1, false, false);
  }

  // D layout: VGPR r, lane l -> M = r + 8*laneHi, N = l&15.
  // Rows 8*laneHi..8*laneHi+7 of this tile are the 8 corners of ONE point, and
  // this lane's (n, p) pair is unique wave-wide -> plain stores, no atomics.
  int n0 = (lane & 15), n1 = 16 + (lane & 15);
  float bv0 = bias[n0], bv1 = bias[n1];
  int prow0 = row0 + (laneHi << 3);         // first of this lane's 8 rows
  int p = prow0 >> 3;                       // the point these rows belong to
  float s0 = 0.f, s1 = 0.f;
  #pragma unroll
  for (int r = 0; r < 8; ++r) {
    int row = prow0 + r;
    float wgt = (row < rows) ? roww[row] : 0.f;
    s0 += (acc0[r] + bv0) * wgt;
    s1 += (acc1[r] + bv1) * wgt;
  }
  if (p < P) {
    out[n0 * P + p] = s0;
    out[n1 * P + p] = s1;
  }
}

extern "C" void kernel_launch(void* const* d_in, const int* in_sizes, int n_in,
                              void* d_out, int out_size, void* d_ws, size_t ws_size,
                              hipStream_t stream) {
  const float* data = (const float*)d_in[0];   // (P,3)
  const float* prev = (const float*)d_in[1];   // (1,32,P)
  const float* w    = (const float*)d_in[2];   // (32,32,3,3,3)
  const float* bias = (const float*)d_in[3];   // (32,)
  // d_in[4] = N (==128, baked into NGRID)
  int P = in_sizes[0] / 3;

  char* ws = (char*)d_ws;
  float*    params  = (float*)(ws + OFF_PARAMS);
  int*      cells   = (int*)(ws + OFF_CELLS);
  float*    splatw  = (float*)(ws + OFF_SPLATW);
  int*      rowcell = (int*)(ws + OFF_ROWCELL);
  float*    roww    = (float*)(ws + OFF_ROWW);
  _Float16* bp      = (_Float16*)(ws + OFF_BPACK);
  float*    grid    = (float*)(ws + OFF_GRID);

  hipMemsetAsync(grid, 0, (size_t)NG3 * CH * sizeof(float), stream);

  k_params<<<1, 256, 0, stream>>>(data, P, params);
  k_point_setup<<<(P + 255) / 256, 256, 0, stream>>>(data, P, params, cells, splatw,
                                                     rowcell, roww);
  k_splat<<<P, 32, 0, stream>>>(prev, P, cells, splatw, grid);
  k_bpack<<<(NTAPS * 2 * 32 * 16 + 255) / 256, 256, 0, stream>>>(w, bp);

  int rows = 8 * P;
  int ntiles = (rows + 15) / 16;
  int wavesPerBlock = 8;                      // 256 threads = 8 wave32
  int nblocks = (ntiles + wavesPerBlock - 1) / wavesPerBlock;
  k_conv_gather<<<nblocks, 32 * wavesPerBlock, 0, stream>>>(grid, bp, bias, rowcell,
                                                            roww, rows, P, (float*)d_out);
}
